// GraphNeuralNetwork_20555713479227
// MI455X (gfx1250) — matrix operand order repaired and checked by reading, not compiled, not run on hardware
//
#include <hip/hip_runtime.h>

// GraphNeuralNetwork as 3x (densify-transposed sparse W -> split f32 into bf16
// hi/lo planes -> pure-bf16 WMMA GEMM (3-term split product) -> bias+ReLU with
// fused re-split of activations for the next layer).
// B=1024, sizes 2048->4096->4096->1024, E=131072 edges per layer.

#define BATCH 1024
#define SZ0 2048
#define SZ1 4096
#define SZ2 4096
#define SZ3 1024

typedef __attribute__((ext_vector_type(16))) __bf16 bf16x16;
typedef __attribute__((ext_vector_type(8)))  __bf16 bf16x8;
typedef __attribute__((ext_vector_type(8)))  float  f32x8;

static __device__ __forceinline__ bf16x16 cat8(bf16x8 a, bf16x8 b) {
  return __builtin_shufflevector(a, b, 0,1,2,3,4,5,6,7,8,9,10,11,12,13,14,15);
}

static __device__ __forceinline__ f32x8 wmma_bf16(bf16x16 a, bf16x16 b, f32x8 c) {
  // (neg_a, A, neg_b, B, c_mod, C, reuse_a, reuse_b)
  return __builtin_amdgcn_wmma_f32_16x16x32_bf16(false, a, false, b, (short)0, c, false, false);
}

// ---------------------------------------------------------------------------
// Plane GEMM:  out = relu(A @ B^T + bias)
//   A planes: [M][K] bf16 (hi,lo);  B planes: [N][K] bf16 (hi,lo)  (W transposed)
// Block: 256 threads (8 waves), tile 128x128, BK=32.
// 3 WMMAs per 16x16 subtile: hi*hi + hi*lo + lo*hi  (~f32 accuracy).
// If outF != null: write f32. Else write bf16 hi/lo planes (next layer's A).
// ---------------------------------------------------------------------------
__global__ __launch_bounds__(256)
void wmma_gemm_planes(const __bf16* __restrict__ Ahi, const __bf16* __restrict__ Alo,
                      const __bf16* __restrict__ Bhi, const __bf16* __restrict__ Blo,
                      const float* __restrict__ bias,
                      float* __restrict__ outF,
                      __bf16* __restrict__ outHi, __bf16* __restrict__ outLo,
                      int N, int K) {
  constexpr int BM = 128, BN = 128, BK = 32, LDL = BK + 8;  // +8 bf16 pad
  __shared__ __align__(16) __bf16 sAhi[BM * LDL];
  __shared__ __align__(16) __bf16 sAlo[BM * LDL];
  __shared__ __align__(16) __bf16 sBhi[BN * LDL];
  __shared__ __align__(16) __bf16 sBlo[BN * LDL];

  const int tid  = threadIdx.x;
  const int lane = tid & 31;
  const int wave = tid >> 5;
  const int wm   = wave >> 2;           // 2 wave-rows -> 64 M-rows each
  const int wn   = wave & 3;            // 4 wave-cols -> 32 N-cols each
  const int m0   = blockIdx.y * BM;
  const int n0   = blockIdx.x * BN;

  // Staging map: each tile plane is 128 rows x 32 bf16 = 512 16B-chunks,
  // 2 chunks per thread (row rr, k-offset cc).
  int rr[2], cc[2];
#pragma unroll
  for (int it = 0; it < 2; ++it) {
    int idx = tid + it * 256;
    rr[it] = idx >> 2;
    cc[it] = (idx & 3) << 3;
  }

  f32x8 c[4][2];
#pragma unroll
  for (int i = 0; i < 4; ++i)
#pragma unroll
    for (int j = 0; j < 2; ++j)
      c[i][j] = (f32x8){0.f,0.f,0.f,0.f,0.f,0.f,0.f,0.f};

  bf16x8 stAh[2], stAl[2], stBh[2], stBl[2];
  auto stage_load = [&](int k0) {
#pragma unroll
    for (int it = 0; it < 2; ++it) {
      size_t aoff = (size_t)(m0 + rr[it]) * K + k0 + cc[it];
      size_t boff = (size_t)(n0 + rr[it]) * K + k0 + cc[it];
      stAh[it] = *(const bf16x8*)(Ahi + aoff);
      stAl[it] = *(const bf16x8*)(Alo + aoff);
      stBh[it] = *(const bf16x8*)(Bhi + boff);
      stBl[it] = *(const bf16x8*)(Blo + boff);
    }
  };

  stage_load(0);

  const int lrow = lane & 15;
  const int akb  = (lane & 16) ? 8 : 0;   // A frag: K kb..kb+7 and kb+16..kb+23
  const int bkb  = (lane & 16) ? 8 : 0;   // B frag: K kb..kb+7 and kb+16..kb+23

  for (int k0 = 0; k0 < K; k0 += BK) {
    __syncthreads();
#pragma unroll
    for (int it = 0; it < 2; ++it) {
      *(bf16x8*)&sAhi[rr[it] * LDL + cc[it]] = stAh[it];
      *(bf16x8*)&sAlo[rr[it] * LDL + cc[it]] = stAl[it];
      *(bf16x8*)&sBhi[rr[it] * LDL + cc[it]] = stBh[it];
      *(bf16x8*)&sBlo[rr[it] * LDL + cc[it]] = stBl[it];
    }
    __syncthreads();
    if (k0 + BK < K) stage_load(k0 + BK);  // prefetch next tile behind WMMAs

    bf16x16 ah[4], al[4], bh[2], bl[2];
#pragma unroll
    for (int i = 0; i < 4; ++i) {
      const __bf16* p = &sAhi[(wm * 64 + i * 16 + lrow) * LDL];
      const __bf16* q = &sAlo[(wm * 64 + i * 16 + lrow) * LDL];
      ah[i] = cat8(*(const bf16x8*)(p + akb), *(const bf16x8*)(p + akb + 16));
      al[i] = cat8(*(const bf16x8*)(q + akb), *(const bf16x8*)(q + akb + 16));
    }
#pragma unroll
    for (int j = 0; j < 2; ++j) {
      const __bf16* p = &sBhi[(wn * 32 + j * 16 + lrow) * LDL];
      const __bf16* q = &sBlo[(wn * 32 + j * 16 + lrow) * LDL];
      bh[j] = cat8(*(const bf16x8*)(p + bkb), *(const bf16x8*)(p + bkb + 16));
      bl[j] = cat8(*(const bf16x8*)(q + bkb), *(const bf16x8*)(q + bkb + 16));
    }
#pragma unroll
    for (int i = 0; i < 4; ++i)
#pragma unroll
      for (int j = 0; j < 2; ++j) {
        c[i][j] = wmma_bf16(ah[i], bh[j], c[i][j]);
        c[i][j] = wmma_bf16(ah[i], bl[j], c[i][j]);
        c[i][j] = wmma_bf16(al[i], bh[j], c[i][j]);
      }
  }

  // Epilogue: bias + ReLU.  C layout: VGPR r -> M=r (lanes<16) / M=r+8.
#pragma unroll
  for (int i = 0; i < 4; ++i) {
    int row0 = m0 + wm * 64 + i * 16 + ((lane & 16) ? 8 : 0);
#pragma unroll
    for (int j = 0; j < 2; ++j) {
      int col = n0 + wn * 32 + j * 16 + (lane & 15);
      float bn = bias[col];
#pragma unroll
      for (int r = 0; r < 8; ++r) {
        float v = c[i][j][r] + bn;
        v = v > 0.f ? v : 0.f;
        size_t off = (size_t)(row0 + r) * N + col;
        if (outF) {
          outF[off] = v;
        } else {
          __bf16 h = (__bf16)v;
          outHi[off] = h;
          outLo[off] = (__bf16)(v - (float)h);
        }
      }
    }
  }
}

// ---------------------------------------------------------------------------
// Support kernels
// ---------------------------------------------------------------------------
__global__ void zero_kernel(float* __restrict__ p, size_t n) {
  size_t i = (size_t)blockIdx.x * blockDim.x + threadIdx.x;
  size_t stride = (size_t)gridDim.x * blockDim.x;
  for (; i < n; i += stride) p[i] = 0.f;
}

// Densify sparse weights TRANSPOSED: WT[dst*K + src] += w  (atomics for dups).
__global__ void build_wT_kernel(const int* __restrict__ src, const int* __restrict__ dst,
                                const float* __restrict__ w, float* __restrict__ WT,
                                int K, int E) {
  int e = blockIdx.x * 256 + threadIdx.x;
  if (e < E) {
    __hip_atomic_fetch_add(&WT[(size_t)dst[e] * K + src[e]], w[e],
                           __ATOMIC_RELAXED, __HIP_MEMORY_SCOPE_AGENT);
  }
}

// f32 -> bf16 hi/lo planes (each element converted exactly once).
__global__ void split_planes_kernel(const float* __restrict__ src,
                                    __bf16* __restrict__ hi, __bf16* __restrict__ lo,
                                    size_t n) {
  size_t i = (size_t)blockIdx.x * blockDim.x + threadIdx.x;
  size_t stride = (size_t)gridDim.x * blockDim.x;
  for (; i < n; i += stride) {
    float f = src[i];
    __bf16 h = (__bf16)f;
    hi[i] = h;
    lo[i] = (__bf16)(f - (float)h);
  }
}

// Fallback (small workspace): direct edge scatter with fp32 atomics.
__global__ void sparse_scatter_kernel(const float* __restrict__ h, const int* __restrict__ src,
                                      const int* __restrict__ dst, const float* __restrict__ w,
                                      float* __restrict__ acc, int K, int N) {
  int e = blockIdx.x;
  int b = blockIdx.y * 256 + threadIdx.x;
  float v = h[(size_t)b * K + src[e]] * w[e];
  __hip_atomic_fetch_add(&acc[(size_t)b * N + dst[e]], v,
                         __ATOMIC_RELAXED, __HIP_MEMORY_SCOPE_AGENT);
}

__global__ void bias_relu_kernel(float* __restrict__ acc, const float* __restrict__ bias,
                                 int N, size_t total) {
  size_t i = (size_t)blockIdx.x * blockDim.x + threadIdx.x;
  size_t stride = (size_t)gridDim.x * blockDim.x;
  for (; i < total; i += stride) {
    float v = acc[i] + bias[i % N];
    acc[i] = v > 0.f ? v : 0.f;
  }
}

// ---------------------------------------------------------------------------
extern "C" void kernel_launch(void* const* d_in, const int* in_sizes, int n_in,
                              void* d_out, int out_size, void* d_ws, size_t ws_size,
                              hipStream_t stream) {
  (void)n_in; (void)out_size;
  const float* x       = (const float*)d_in[0];
  const int*   src[3]  = {(const int*)d_in[1], (const int*)d_in[5], (const int*)d_in[9]};
  const int*   dst[3]  = {(const int*)d_in[2], (const int*)d_in[6], (const int*)d_in[10]};
  const float* wts[3]  = {(const float*)d_in[3], (const float*)d_in[7], (const float*)d_in[11]};
  const float* bias[3] = {(const float*)d_in[4], (const float*)d_in[8], (const float*)d_in[12]};
  const int E = in_sizes[1];

  const int Ks[3] = {SZ0, SZ1, SZ2};
  const int Ns[3] = {SZ1, SZ2, SZ3};

  // Workspace layout (plane path):
  //   WT f32 (4096*4096) | WThi | WTlo | xhi | xlo | h1hi | h1lo | h2hi | h2lo
  const size_t WCAP = (size_t)4096 * 4096;
  float*  WT   = (float*)d_ws;
  __bf16* WThi = (__bf16*)(WT + WCAP);
  __bf16* WTlo = WThi + WCAP;
  __bf16* xhi  = WTlo + WCAP;
  __bf16* xlo  = xhi + (size_t)BATCH * SZ0;
  __bf16* h1hi = xlo + (size_t)BATCH * SZ0;
  __bf16* h1lo = h1hi + (size_t)BATCH * SZ1;
  __bf16* h2hi = h1lo + (size_t)BATCH * SZ1;
  __bf16* h2lo = h2hi + (size_t)BATCH * SZ2;
  const size_t need = (size_t)((char*)(h2lo + (size_t)BATCH * SZ2) - (char*)d_ws);

  if (ws_size >= need) {
    // Split the network input once.
    split_planes_kernel<<<4096, 256, 0, stream>>>(x, xhi, xlo, (size_t)BATCH * SZ0);

    const __bf16* Ah[3] = {xhi, h1hi, h2hi};
    const __bf16* Al[3] = {xlo, h1lo, h2lo};
    __bf16* oHi[3] = {h1hi, h2hi, nullptr};
    __bf16* oLo[3] = {h1lo, h2lo, nullptr};
    float*  oF [3] = {nullptr, nullptr, (float*)d_out};

    for (int l = 0; l < 3; ++l) {
      const size_t wcount = (size_t)Ks[l] * Ns[l];
      zero_kernel<<<4096, 256, 0, stream>>>(WT, wcount);
      build_wT_kernel<<<(E + 255) / 256, 256, 0, stream>>>(src[l], dst[l], wts[l], WT, Ks[l], E);
      split_planes_kernel<<<4096, 256, 0, stream>>>(WT, WThi, WTlo, wcount);
      dim3 grid(Ns[l] / 128, BATCH / 128);
      wmma_gemm_planes<<<grid, 256, 0, stream>>>(Ah[l], Al[l], WThi, WTlo, bias[l],
                                                 oF[l], oHi[l], oLo[l], Ns[l], Ks[l]);
    }
  } else {
    // Small-workspace fallback: direct sparse scatter (needs 32 MB).
    float* f1 = (float*)d_ws;
    float* f2 = f1 + (size_t)BATCH * 4096;
    if (ws_size < 2 * (size_t)BATCH * 4096 * sizeof(float)) return;
    const float* in = x;
    float* outs[3] = {f1, f2, (float*)d_out};
    for (int l = 0; l < 3; ++l) {
      const size_t total = (size_t)BATCH * Ns[l];
      zero_kernel<<<4096, 256, 0, stream>>>(outs[l], total);
      dim3 grid(E, BATCH / 256);
      sparse_scatter_kernel<<<grid, 256, 0, stream>>>(in, src[l], dst[l], wts[l], outs[l], Ks[l], Ns[l]);
      bias_relu_kernel<<<2048, 256, 0, stream>>>(outs[l], bias[l], Ns[l], total);
      in = outs[l];
    }
  }
}
